// QNet_87960930222359
// MI455X (gfx1250) — matrix operand order repaired and checked by reading, not compiled, not run on hardware
//
#include <hip/hip_runtime.h>

typedef __attribute__((ext_vector_type(16))) _Float16 v16h;
typedef __attribute__((ext_vector_type(8)))  _Float16 v8h;
typedef __attribute__((ext_vector_type(8)))  float    v8f;
typedef __attribute__((ext_vector_type(4)))  unsigned int v4u;
typedef __attribute__((ext_vector_type(8)))  int      v8i;
typedef __attribute__((ext_vector_type(4)))  int      v4i;

// ---------------------------------------------------------------------------
// Kernel 1: per-image workgroup (128 thr = 4 waves).
//   TDM-stage x[b] -> LDS (f32), convert to f16,
//   conv1 as implicit GEMM on WMMA f16 (K=9 zero-padded to 32, N=6 pad 16),
//   relu + maxpool -> h1s,
//   conv2 as implicit GEMM on WMMA f16 (K=54 zero-padded to 64, N=16 exact),
//   relu + maxpool -> h2[b][576] f16 (flatten order oc*36 + py*6 + px).
// All WMMA-feeding gathers/stores are branch-free (zero-padded LDS + clamps).
// ---------------------------------------------------------------------------
__global__ __launch_bounds__(128)
void qnet_conv_front(const float* __restrict__ x,
                     const float* __restrict__ w1, const float* __restrict__ b1,
                     const float* __restrict__ w2, const float* __restrict__ b2,
                     _Float16* __restrict__ h2)
{
    __shared__ __align__(16) float    xs[32 * 32];     //  4.0 KB staged image f32
    __shared__ __align__(16) _Float16 xsh[1040];       //  2.1 KB f16 image, [1024]=0
    __shared__ __align__(16) float    c1s[6 * 912];    // 21.9 KB conv1 out (900->912)
    __shared__ __align__(16) _Float16 h1s[8 * 225];    //  3.6 KB (ic 6,7 = zero pad)
    __shared__ __align__(16) float    c2s[16 * 176];   // 11.3 KB (pos 169->176)

    const int b    = blockIdx.x;
    const int tid  = threadIdx.x;
    const int lane = tid & 31;
    const int wave = tid >> 5;

    const float* xb = x + (size_t)b * 1024;

    // ---- zero pad regions ----
    for (int i = tid; i < 2 * 225; i += 128) h1s[6 * 225 + i] = (_Float16)0.f;
    if (tid == 0) xsh[1024] = (_Float16)0.f;

    // ---- stage input image into LDS ----
#if __has_builtin(__builtin_amdgcn_tensor_load_to_lds) && \
    __has_builtin(__builtin_amdgcn_s_wait_tensorcnt)
    if (wave == 0) {
        // Tensor DMA descriptor (ISA 8.3/8.4): 1D tile, 1024 x 4-byte elements.
        const unsigned long long ga = (unsigned long long)(uintptr_t)xb;
        const unsigned int lds_off  = (unsigned int)(uintptr_t)(&xs[0]);
        v4u g0 = { 1u,                               // count=1 user descriptor
                   lds_off,                          // lds_addr
                   (unsigned int)ga,                 // global_addr[31:0]
                   (unsigned int)((ga >> 32) & 0x01FFFFFFull) | (2u << 30) }; // addr[56:32], type=2
        v8i g1 = { (int)(2u << 16),                  // data_size = 4B (code 2)
                   (int)(1024u << 16),               // tensor_dim0 = 1024
                   (int)(1u << 16),                  // tensor_dim1 = 1
                   (int)(1024u << 16),               // tile_dim0 = 1024
                   0,                                // tile_dim1/2 unused
                   1024,                             // tensor_dim0_stride
                   0, 0 };
        v4i gz = { 0, 0, 0, 0 };
#if defined(__clang_major__) && (__clang_major__ >= 23)
        v8i gz8 = { 0, 0, 0, 0, 0, 0, 0, 0 };
        __builtin_amdgcn_tensor_load_to_lds(g0, g1, gz, gz, gz8, 0);
#else
        __builtin_amdgcn_tensor_load_to_lds(g0, g1, gz, gz, 0);
#endif
        __builtin_amdgcn_s_wait_tensorcnt(0);
    }
#else
    for (int i = tid; i < 1024; i += 128) xs[i] = xb[i];
#endif
    __syncthreads();

    // ---- convert image to f16 for WMMA A-gathers ----
    for (int i = tid; i < 1024; i += 128) xsh[i] = (_Float16)xs[i];
    __syncthreads();

    // A-layout element -> K (ISA 7.12.2): lanes 0-15: e<8 -> K=e, e>=8 -> K=e+8
    //                                     lanes16-31: +8.
    const int khalf = (lane & 16) ? 8 : 0;

    // ================= conv1 GEMM: C[900x6] = P[900x9] * W1^T[9x6] ==========
    // K padded 9->32 (one f16 WMMA chunk), N padded 6->16.
    {
        // B fragment (weights), one-time: N = lane&15, e -> K = e + 8*lane[4] + (e>=8 ? 8 : 0)
        v16h bw1;
        {
            const int n  = lane & 15;
            const int kb = (lane & 16) ? 8 : 0;
            #pragma unroll
            for (int e = 0; e < 16; ++e) {
                const int k = kb + e + ((e >= 8) ? 8 : 0);
                bw1[e] = (n < 6 && k < 9) ? (_Float16)w1[n * 9 + k]
                                          : (_Float16)0.f;
            }
        }
        const float bias1 = b1[(lane & 15) < 6 ? (lane & 15) : 0];

        // tile-invariant element offsets; invalid K -> sentinel 1024 (zero elt)
        int eoffX[16];
        #pragma unroll
        for (int e = 0; e < 16; ++e) {
            const int k = e + khalf + ((e >= 8) ? 8 : 0);   // K: 0..31
            eoffX[e] = (k < 9) ? (k / 3) * 32 + (k % 3) : 1024;
        }

        // 900 positions -> 57 M-tiles of 16; waves stride over tiles.
        const int oc  = lane & 15;
        const bool cv = (oc < 6);
        for (int t = wave; t < 57; t += 4) {
            const int p  = min(t * 16 + (lane & 15), 899);  // clamped garbage rows
            const int oy = p / 30, ox = p % 30;             // land in c1s padding
            const int base = oy * 32 + ox;
            v16h af;
            #pragma unroll
            for (int e = 0; e < 16; ++e)
                af[e] = xsh[min(base + eoffX[e], 1024)];
            v8f c = {};
            c = __builtin_amdgcn_wmma_f32_16x16x32_f16(false, af, false, bw1,
                                                       (short)0, c, false, false);
            // C layout: VGPR r -> M = r + 8*lane[4], N = lane&15 (= channel)
            if (cv) {                                       // single per-lane guard
                float* crow = &c1s[oc * 912 + t * 16 + ((lane & 16) ? 8 : 0)];
                #pragma unroll
                for (int r = 0; r < 8; ++r)
                    crow[r] = fmaxf(c[r] + bias1, 0.f);
            }
        }
    }
    __syncthreads();

    // ---- maxpool conv1: 30x30 -> 15x15, store f16 -> h1s[6][15][15] ----
    for (int idx = tid; idx < 6 * 225; idx += 128) {
        const int oc = idx / 225, r = idx % 225;
        const int py = r / 15, px = r % 15;
        const float* cc = &c1s[oc * 912];
        const float m0 = fmaxf(cc[(2 * py) * 30 + 2 * px],
                               cc[(2 * py) * 30 + 2 * px + 1]);
        const float m1 = fmaxf(cc[(2 * py + 1) * 30 + 2 * px],
                               cc[(2 * py + 1) * 30 + 2 * px + 1]);
        h1s[idx] = (_Float16)fmaxf(m0, m1);
    }
    __syncthreads();

    // ================ conv2 GEMM: C[169x16] = P[169x54] * W2^T[54x16] =======
    // B fragments (weights), K padded 54 -> 64.
    v16h bf0, bf1;
    {
        const int n  = lane & 15;
        const int kb = (lane & 16) ? 8 : 0;
        #pragma unroll
        for (int e = 0; e < 16; ++e) {
            const int k0 = kb + e + ((e >= 8) ? 8 : 0); // 0..31, always < 54
            const int k1 = k0 + 32;         // 32..63
            bf0[e] = (_Float16)w2[n * 54 + k0];
            bf1[e] = (k1 < 54) ? (_Float16)w2[n * 54 + k1] : (_Float16)0.f;
        }
    }
    const float bias2 = b2[lane & 15];

    // tile-invariant offsets; k>=54 lands in the zeroed ic=6/7 pad region.
    int eoffA0[16], eoffA1[16];
    #pragma unroll
    for (int e = 0; e < 16; ++e) {
        const int k  = e + khalf + ((e >= 8) ? 8 : 0); // chunk0 K: 0..31
        const int i0 = k / 9,  r0 = k % 9;
        eoffA0[e] = i0 * 225 + (r0 / 3) * 15 + (r0 % 3);
        const int k2 = k + 32;                         // chunk1 K: 32..63
        const int i1 = k2 / 9, r1 = k2 % 9;
        eoffA1[e] = i1 * 225 + (r1 / 3) * 15 + (r1 % 3);
    }

    // 169 output positions -> 11 M-tiles of 16; waves stride over tiles.
    for (int t = wave; t < 11; t += 4) {
        const int p  = min(t * 16 + (lane & 15), 168); // clamp: garbage rows are
        const int oy = p / 13, ox = p % 13;            // discarded via c2s padding
        const int base = oy * 15 + ox;
        v16h af0, af1;
        #pragma unroll
        for (int e = 0; e < 16; ++e) {
            af0[e] = h1s[base + eoffA0[e]];
            af1[e] = h1s[base + eoffA1[e]];
        }
        v8f c = {};
        c = __builtin_amdgcn_wmma_f32_16x16x32_f16(false, af0, false, bf0,
                                                   (short)0, c, false, false);
        c = __builtin_amdgcn_wmma_f32_16x16x32_f16(false, af1, false, bf1,
                                                   (short)0, c, false, false);
        // C layout: VGPR r -> M = r + 8*lane[4], N = lane&15 (= out channel)
        const int oc  = lane & 15;
        const int pp0 = t * 16 + ((lane & 16) ? 8 : 0);
        float* crow = &c2s[oc * 176 + pp0];
        #pragma unroll
        for (int r = 0; r < 8; ++r)
            crow[r] = fmaxf(c[r] + bias2, 0.f);        // unconditional, padded
    }
    __syncthreads();

    // ---- maxpool 13x13 -> 6x6, flatten [16,6,6] -> 576, store f16 ----
    _Float16* hrow = h2 + (size_t)b * 576;
    for (int idx = tid; idx < 576; idx += 128) {
        const int oc = idx / 36, r = idx % 36, py = r / 6, px = r % 6;
        const float* cc = &c2s[oc * 176];
        const float m0 = fmaxf(cc[(2 * py) * 13 + 2 * px],
                               cc[(2 * py) * 13 + 2 * px + 1]);
        const float m1 = fmaxf(cc[(2 * py + 1) * 13 + 2 * px],
                               cc[(2 * py + 1) * 13 + 2 * px + 1]);
        hrow[idx] = (_Float16)fmaxf(m0, m1);
    }
}

// ---------------------------------------------------------------------------
// Kernel 2: fc1 GEMM [B,576] x [576,4->16] via WMMA, relu -> angles[B][4] f32.
// One wave per 16-row M-tile; 18 k-chunks of 32.  The zero-padded B matrix is
// staged once per block into LDS in per-lane *fragment order*, so the k-loop
// is two 16B global loads + one 32B LDS load + one WMMA, branch-free.
// ---------------------------------------------------------------------------
__global__ __launch_bounds__(256)
void qnet_fc1(const _Float16* __restrict__ h2,
              const float* __restrict__ fc1_w, const float* __restrict__ fc1_b,
              float* __restrict__ angles, int ntiles)
{
    __shared__ __align__(32) _Float16 Wfrag[18 * 32 * 16];   // 18 KB

    const int tid  = threadIdx.x;
    const int lane = tid & 31;
    const int wave = tid >> 5;
    const int tile = blockIdx.x * 8 + wave;

    // Fill B fragments: Wfrag[kc][lane][e] = B[kc*32 + 8*lane[4] + e + (e>=8 ? 8 : 0)][lane&15]
    for (int idx = tid; idx < 18 * 512; idx += 256) {
        const int kc = idx >> 9;
        const int ln = (idx >> 4) & 31;
        const int e  = idx & 15;
        const int n  = ln & 15;
        const int k  = kc * 32 + ((ln & 16) ? 8 : 0) + e + ((e >= 8) ? 8 : 0);
        Wfrag[idx] = (n < 4) ? (_Float16)fc1_w[n * 576 + k] : (_Float16)0.f;
    }
    __syncthreads();

    if (tile < ntiles) {                       // wave-uniform; EXEC all-1s inside
        const int n   = lane & 15;             // output column (padded N)
        const int row = tile * 16 + (lane & 15);
        const _Float16* arow = h2 + (size_t)row * 576;
        const int aoff0 = (lane & 16) ? 8 : 0; // K of element 0 (A layout)
        const int aoff1 = aoff0 + 16;          // K of element 8
        const _Float16* wl = Wfrag + (lane << 4);

        v8f c = {};
        for (int kc = 0; kc < 18; ++kc) {
            const int kbase = kc * 32;
            const v8h lo = *(const v8h*)(arow + kbase + aoff0);
            const v8h hi = *(const v8h*)(arow + kbase + aoff1);
            const v16h a = __builtin_shufflevector(lo, hi,
                0, 1, 2, 3, 4, 5, 6, 7, 8, 9, 10, 11, 12, 13, 14, 15);
            const v16h bfr = *(const v16h*)(wl + (kc << 9));
            c = __builtin_amdgcn_wmma_f32_16x16x32_f16(false, a, false, bfr,
                                                       (short)0, c, false, false);
        }
        if (n < 4) {
            const float bias = fc1_b[n];
            #pragma unroll
            for (int r = 0; r < 8; ++r) {
                const int brow = tile * 16 + r + ((lane & 16) ? 8 : 0);
                angles[brow * 4 + n] = fmaxf(c[r] + bias, 0.f);
            }
        }
    }
}

// ---------------------------------------------------------------------------
// Kernel 3: 4-qubit statevector (16 complex amps in VGPRs, one item/lane),
// 6 layers of RX + ring CNOTs, <Z_w> expectations, then fc2+relu and fc3.
// wire w maps to flat-index bit (3-w): idx = i_q0*8 + i_q1*4 + i_q2*2 + i_q3.
// ---------------------------------------------------------------------------
__global__ __launch_bounds__(256)
void qnet_quantum_tail(const float* __restrict__ angles,
                       const float* __restrict__ qw,
                       const float* __restrict__ fc2_w, const float* __restrict__ fc2_b,
                       const float* __restrict__ fc3_w, const float* __restrict__ fc3_b,
                       float* __restrict__ out, int B)
{
    const int b = blockIdx.x * blockDim.x + threadIdx.x;
    if (b >= B) return;

    float cw[4], sw[4];
    #pragma unroll
    for (int w = 0; w < 4; ++w) {
        const float h = angles[b * 4 + w] * 0.5f;
        cw[w] = __cosf(h);
        sw[w] = __sinf(h);
    }
    // product state: factor is cos (bit=0) or -i*sin (bit=1)
    float cre[16], cim[16];
    #pragma unroll
    for (int i = 0; i < 16; ++i) {
        float re = 1.f, im = 0.f;
        #pragma unroll
        for (int w = 0; w < 4; ++w) {
            if ((i >> (3 - w)) & 1) {      // *(-i*s): (re,im)->(im*s, -re*s)
                const float nr = im * sw[w], ni = -re * sw[w];
                re = nr; im = ni;
            } else {
                re *= cw[w]; im *= cw[w];
            }
        }
        cre[i] = re; cim[i] = im;
    }
    // BasicEntanglerLayers
    #pragma unroll
    for (int l = 0; l < 6; ++l) {
        #pragma unroll
        for (int w = 0; w < 4; ++w) {
            const float th = qw[l * 4 + w] * 0.5f;
            const float ct = __cosf(th), sn = __sinf(th);
            const int s = 8 >> w;
            #pragma unroll
            for (int i = 0; i < 16; ++i) {
                if (i & s) continue;
                const int j = i | s;
                const float a0r = cre[i], a0i = cim[i];
                const float a1r = cre[j], a1i = cim[j];
                cre[i] = ct * a0r + sn * a1i;  cim[i] = ct * a0i - sn * a1r;
                cre[j] = ct * a1r + sn * a0i;  cim[j] = ct * a1i - sn * a0r;
            }
        }
        #pragma unroll
        for (int e = 0; e < 4; ++e) {       // ring CNOTs (0,1)(1,2)(2,3)(3,0)
            const int sc  = 8 >> e;
            const int stt = 8 >> ((e + 1) & 3);
            #pragma unroll
            for (int i = 0; i < 16; ++i) {
                if ((i & sc) && !(i & stt)) {
                    const int j = i | stt;
                    const float tr = cre[i], ti = cim[i];
                    cre[i] = cre[j]; cim[i] = cim[j];
                    cre[j] = tr;     cim[j] = ti;
                }
            }
        }
    }
    // <Z_w>
    float ev[4];
    #pragma unroll
    for (int w = 0; w < 4; ++w) {
        const int mask = 8 >> w;
        float s = 0.f;
        #pragma unroll
        for (int i = 0; i < 16; ++i) {
            const float p = cre[i] * cre[i] + cim[i] * cim[i];
            s += (i & mask) ? -p : p;
        }
        ev[w] = s;
    }
    // fc2 (4->32) + relu, fc3 (32->10)
    float y[32];
    #pragma unroll
    for (int j = 0; j < 32; ++j) {
        float acc = fc2_b[j];
        #pragma unroll
        for (int n = 0; n < 4; ++n) acc += fc2_w[j * 4 + n] * ev[n];
        y[j] = fmaxf(acc, 0.f);
    }
    #pragma unroll
    for (int k = 0; k < 10; ++k) {
        float acc = fc3_b[k];
        #pragma unroll
        for (int j = 0; j < 32; ++j) acc += fc3_w[k * 32 + j] * y[j];
        out[b * 10 + k] = acc;
    }
}

// ---------------------------------------------------------------------------
extern "C" void kernel_launch(void* const* d_in, const int* in_sizes, int n_in,
                              void* d_out, int out_size, void* d_ws, size_t ws_size,
                              hipStream_t stream) {
    const float* x      = (const float*)d_in[0];
    const float* w1     = (const float*)d_in[1];
    const float* b1     = (const float*)d_in[2];
    const float* w2     = (const float*)d_in[3];
    const float* b2     = (const float*)d_in[4];
    const float* fc1_w  = (const float*)d_in[5];
    const float* fc1_b  = (const float*)d_in[6];
    const float* qw     = (const float*)d_in[7];
    const float* fc2_w  = (const float*)d_in[8];
    const float* fc2_b  = (const float*)d_in[9];
    const float* fc3_w  = (const float*)d_in[10];
    const float* fc3_b  = (const float*)d_in[11];
    float* out = (float*)d_out;

    const int B = in_sizes[0] / 1024;       // x is [B,1,32,32]

    // workspace layout: h2 f16 [B,576] then angles f32 [B,4]
    _Float16* h2 = (_Float16*)d_ws;
    float* angles = (float*)((char*)d_ws + (size_t)B * 576 * sizeof(_Float16));

    qnet_conv_front<<<B, 128, 0, stream>>>(x, w1, b1, w2, b2, h2);

    const int ntiles = B / 16;              // 512 for B=8192
    qnet_fc1<<<(ntiles + 7) / 8, 256, 0, stream>>>(h2, fc1_w, fc1_b, angles, ntiles);

    qnet_quantum_tail<<<(B + 255) / 256, 256, 0, stream>>>(
        angles, qw, fc2_w, fc2_b, fc3_w, fc3_b, out, B);
}
